// TransducerFast_32830730010925
// MI455X (gfx1250) — matrix operand order repaired and checked by reading, not compiled, not run on hardware
//
#include <hip/hip_runtime.h>
#include <hip/hip_bf16.h>
#include <stdint.h>

// ---------------- problem constants ----------------
#define BB   8
#define TT   512
#define SS   64
#define VV   4233
#define DD   256
#define JJ   512
#define RR   5
#define SP1  (SS + 1)          // 65
#define VP   4256              // V padded to multiple of 32 (WMMA K chunks)
#define MT   (BB * TT)         // 4096 encoder rows
#define MPRED  (BB * SP1)      // 520 predictor rows
#define MPREDP 528             // padded to multiple of 16
#define NEGV (-1e30f)
#define NTILES_V ((VV + 15) / 16)   // 265 N tiles over vocab
#define USE_TDM 1

typedef __bf16 bf16_t;
typedef __attribute__((ext_vector_type(16))) __bf16 v16bf;
typedef __attribute__((ext_vector_type(8)))  __bf16 v8bf;
typedef __attribute__((ext_vector_type(8)))  float  v8f;
typedef __attribute__((ext_vector_type(4)))  unsigned int v4u;
typedef __attribute__((ext_vector_type(8)))  int v8i;

// ---------------- helpers ----------------
__device__ __forceinline__ bf16_t f2b(float f) {
    union { float f; unsigned u; } in; in.f = f;
    unsigned u = in.u;
    unsigned r = u + 0x7FFFu + ((u >> 16) & 1u);     // RNE
    union { unsigned short s; bf16_t b; } out; out.s = (unsigned short)(r >> 16);
    return out.b;
}

__device__ __forceinline__ float ladd(float a, float b) {
    float m = fmaxf(a, b);
    if (m <= -1e29f) return NEGV;
    float lo = fminf(a, b);
    return m + log1pf(expf(lo - m));
}

// A fragment: 16x32 bf16, lane l -> row (l&15), K chunks at k0+(l>>4)*8 and +16
__device__ __forceinline__ v16bf load_frag_a(const bf16_t* A, int lda, int lane, int k0) {
    const bf16_t* p = A + (size_t)(lane & 15) * lda + k0 + ((lane >> 4) << 3);
    v8bf lo = *(const v8bf*)p;
    v8bf hi = *(const v8bf*)(p + 16);
    v16bf r;
#pragma unroll
    for (int i = 0; i < 8; ++i) { r[i] = lo[i]; r[i + 8] = hi[i]; }
    return r;
}

// B fragment from row-major W[N][K] (i.e. matrix B = W^T):
// lane l -> column (l&15), K chunks at k0+(l>>4)*8 and +16
__device__ __forceinline__ v16bf load_frag_b(const bf16_t* W, int ldw, int lane, int k0) {
    const bf16_t* p = W + (size_t)(lane & 15) * ldw + k0 + ((lane >> 4) << 3);
    v8bf lo = *(const v8bf*)p;
    v8bf hi = *(const v8bf*)(p + 16);
    v16bf r;
#pragma unroll
    for (int i = 0; i < 8; ++i) { r[i] = lo[i]; r[i + 8] = hi[i]; }
    return r;
}

// ---------------- small prep kernels ----------------
__global__ void k_scale_w_bf(const float* __restrict__ src, bf16_t* __restrict__ dst,
                             int N0, int Np, int K, const float* __restrict__ ws) {
    size_t i = (size_t)blockIdx.x * blockDim.x + threadIdx.x;
    if (i >= (size_t)Np * K) return;
    int n = (int)(i / K), k = (int)(i % K);
    float s = ws ? expf(ws[0]) : 1.0f;
    float v = (n < N0) ? src[(size_t)n * K + k] * s : 0.0f;
    dst[i] = f2b(v);
}

__global__ void k_scale_bias(const float* __restrict__ src, float* __restrict__ dst,
                             int N0, int Np, const float* __restrict__ bs) {
    int i = blockIdx.x * blockDim.x + threadIdx.x;
    if (i >= Np) return;
    float s = bs ? expf(bs[0]) : 1.0f;
    dst[i] = (i < N0) ? src[i] * s : 0.0f;
}

__global__ void k_cvt_enc(const float* __restrict__ src, bf16_t* __restrict__ dst) {
    size_t i = (size_t)blockIdx.x * blockDim.x + threadIdx.x;
    if (i >= (size_t)MT * DD) return;
    dst[i] = f2b(src[i]);
}

__global__ void k_gather_pred(const int* __restrict__ y, const float* __restrict__ emb,
                              float* __restrict__ pf, bf16_t* __restrict__ pb) {
    size_t i = (size_t)blockIdx.x * blockDim.x + threadIdx.x;
    if (i >= (size_t)MPREDP * DD) return;
    int m = (int)(i / DD), k = (int)(i % DD);
    float v = 0.0f;
    if (m < MPRED) {
        int b = m / SP1, s = m % SP1;
        int tok = (s == 0) ? 0 : y[b * SS + s - 1];   // BLANK-prefixed
        v = emb[(size_t)tok * DD + k];
    }
    pf[i] = v;
    pb[i] = f2b(v);
}

__global__ void k_fill(float* __restrict__ p, size_t n, float val) {
    size_t i = (size_t)blockIdx.x * blockDim.x + threadIdx.x;
    if (i < n) p[i] = val;
}

// ---------------- generic WMMA GEMM: C[M,N] = A[M,K] * W[N,K]^T + bias ----------------
__global__ void __launch_bounds__(32)
k_gemm_bf16(const bf16_t* __restrict__ A, const bf16_t* __restrict__ W,
            const float* __restrict__ bias, float* __restrict__ C,
            int M, int N, int K, int lda, int ldw, int ldc) {
    int tm = blockIdx.x, tn = blockIdx.y;
    int lane = threadIdx.x & 31;
    const bf16_t* Ab = A + (size_t)tm * 16 * lda;
    const bf16_t* Wb = W + (size_t)tn * 16 * ldw;
    v8f c = {};
    for (int k0 = 0; k0 < K; k0 += 32) {
        __builtin_prefetch((const void*)(Wb + (size_t)(lane & 15) * ldw + k0 + 128), 0, 1);
        v16bf a = load_frag_a(Ab, lda, lane, k0);
        v16bf b = load_frag_b(Wb, ldw, lane, k0);
        c = __builtin_amdgcn_wmma_f32_16x16x32_bf16(false, a, false, b, (short)0, c, false, false);
    }
    int col = tn * 16 + (lane & 15);
    if (col < N) {
        float bv = bias ? bias[col] : 0.0f;
#pragma unroll
        for (int j = 0; j < 8; ++j) {
            int row = tm * 16 + j + ((lane >> 4) << 3);
            if (row < M) C[(size_t)row * ldc + col] = c[j] + bv;
        }
    }
}

// ---------------- row max over first `cols` columns ----------------
__global__ void __launch_bounds__(32)
k_rowmax(const float* __restrict__ X, int ld, int cols, float* __restrict__ out) {
    int r = blockIdx.x;
    int lane = threadIdx.x;
    const float* p = X + (size_t)r * ld;
    float m = -3.0e38f;
    for (int v = lane; v < cols; v += 32) m = fmaxf(m, p[v]);
#pragma unroll
    for (int off = 16; off > 0; off >>= 1) m = fmaxf(m, __shfl_xor(m, off, 32));
    if (lane == 0) out[r] = m;
}

__global__ void k_exp_am(const float* __restrict__ am, const float* __restrict__ amax,
                         bf16_t* __restrict__ eam) {
    size_t i = (size_t)blockIdx.x * blockDim.x + threadIdx.x;
    if (i >= (size_t)MT * VP) return;
    int m = (int)(i / VP), v = (int)(i % VP);
    float e = 0.0f;
    if (v < VV) e = expf(am[i] - amax[m]);
    eam[i] = f2b(e);
}

__global__ void k_exp_lm(const float* __restrict__ lm, const float* __restrict__ lmax,
                         bf16_t* __restrict__ elm) {
    size_t i = (size_t)blockIdx.x * blockDim.x + threadIdx.x;
    if (i >= (size_t)BB * 80 * VP) return;
    int v = (int)(i % VP);
    size_t r = i / VP;
    int s = (int)(r % 80), b = (int)(r / 80);
    float e = 0.0f;
    if (s < SP1 && v < VV) e = expf(lm[((size_t)(b * SP1 + s)) * VP + v] - lmax[b * SP1 + s]);
    elm[i] = f2b(e);
}

// ---------------- norm[b,s,t] = log(e_lm[s,:] . e_am[t,:] + 1e-30) + lmax + amax ----------------
__global__ void __launch_bounds__(32)
k_gemm_norm(const bf16_t* __restrict__ elm, const bf16_t* __restrict__ eam,
            const float* __restrict__ lmax, const float* __restrict__ amax,
            float* __restrict__ nrm) {
    int tm = blockIdx.x, tn = blockIdx.y, b = blockIdx.z;
    int lane = threadIdx.x & 31;
    const bf16_t* Ab = elm + (size_t)b * 80 * VP + (size_t)tm * 16 * VP;
    const bf16_t* Wb = eam + (size_t)b * TT * VP + (size_t)tn * 16 * VP;
    v8f c = {};
    for (int k0 = 0; k0 < VP; k0 += 32) {
        v16bf a = load_frag_a(Ab, VP, lane, k0);
        v16bf w = load_frag_b(Wb, VP, lane, k0);
        c = __builtin_amdgcn_wmma_f32_16x16x32_bf16(false, a, false, w, (short)0, c, false, false);
    }
    int t = tn * 16 + (lane & 15);
#pragma unroll
    for (int j = 0; j < 8; ++j) {
        int s = tm * 16 + j + ((lane >> 4) << 3);
        if (s < SP1)
            nrm[((size_t)(b * 80 + s)) * TT + t] =
                logf(c[j] + 1e-30f) + lmax[b * SP1 + s] + amax[b * TT + t];
    }
}

// ---------------- px / py (stage 1) ----------------
__global__ void k_build_px(const float* __restrict__ am, const float* __restrict__ lm,
                           const float* __restrict__ nrm, const int* __restrict__ y,
                           float* __restrict__ px) {
    size_t i = (size_t)blockIdx.x * blockDim.x + threadIdx.x;
    if (i >= (size_t)BB * SS * (TT + 1)) return;
    int t = (int)(i % (TT + 1));
    size_t r = i / (TT + 1);
    int s = (int)(r % SS), b = (int)(r / SS);
    if (t == TT) { px[i] = NEGV; return; }
    int lab = y[b * SS + s];
    px[i] = am[((size_t)(b * TT + t)) * VP + lab] +
            lm[((size_t)(b * SP1 + s)) * VP + lab] -
            nrm[((size_t)(b * 80 + s)) * TT + t];
}

__global__ void k_build_py(const float* __restrict__ am, const float* __restrict__ lm,
                           const float* __restrict__ nrm, float* __restrict__ py) {
    size_t i = (size_t)blockIdx.x * blockDim.x + threadIdx.x;
    if (i >= (size_t)BB * SP1 * TT) return;
    int t = (int)(i % TT);
    size_t r = i / TT;
    int s = (int)(r % SP1), b = (int)(r / SP1);
    py[i] = am[((size_t)(b * TT + t)) * VP + 0] +
            lm[((size_t)(b * SP1 + s)) * VP + 0] -
            nrm[((size_t)(b * 80 + s)) * TT + t];
}

// ---------------- mutual-information lattice: forward alpha ----------------
__global__ void k_mi_forward(const float* __restrict__ px, const float* __restrict__ py,
                             float* __restrict__ al, float* __restrict__ loss) {
    int b = blockIdx.x;
    float* A = al + (size_t)b * SP1 * (TT + 1);
    const float* PX = px + (size_t)b * SS * (TT + 1);
    const float* PY = py + (size_t)b * SP1 * TT;
    for (int d = 0; d <= SS + TT; ++d) {
        for (int s = threadIdx.x; s <= SS; s += blockDim.x) {
            int t = d - s;
            if (t >= 0 && t <= TT) {
                float v;
                if (s == 0 && t == 0) v = 0.0f;
                else {
                    float v1 = (s > 0) ? A[(s - 1) * (TT + 1) + t] + PX[(s - 1) * (TT + 1) + t] : NEGV;
                    float v2 = (t > 0) ? A[s * (TT + 1) + t - 1] + PY[s * TT + (t - 1)] : NEGV;
                    v = ladd(v1, v2);
                }
                A[s * (TT + 1) + t] = v;
            }
        }
        __syncthreads();
    }
    if (threadIdx.x == 0) loss[b] = A[SS * (TT + 1) + TT];
}

// ---------------- occupancy (adjoint of alpha) ----------------
__global__ void k_mi_occ(const float* __restrict__ px, const float* __restrict__ py,
                         const float* __restrict__ al, float* __restrict__ occ) {
    int b = blockIdx.x;
    const float* A = al + (size_t)b * SP1 * (TT + 1);
    float* G = occ + (size_t)b * SP1 * (TT + 1);
    const float* PX = px + (size_t)b * SS * (TT + 1);
    const float* PY = py + (size_t)b * SP1 * TT;
    for (int d = SS + TT; d >= 0; --d) {
        for (int s = threadIdx.x; s <= SS; s += blockDim.x) {
            int t = d - s;
            if (t >= 0 && t <= TT) {
                float g;
                if (s == SS && t == TT) g = 1.0f;
                else {
                    float a = A[s * (TT + 1) + t];
                    g = 0.0f;
                    if (s < SS) {
                        float arg = a + PX[s * (TT + 1) + t] - A[(s + 1) * (TT + 1) + t];
                        g += G[(s + 1) * (TT + 1) + t] * expf(fminf(arg, 30.0f));
                    }
                    if (t < TT) {
                        float arg = a + PY[s * TT + t] - A[s * (TT + 1) + t + 1];
                        g += G[s * (TT + 1) + t + 1] * expf(fminf(arg, 30.0f));
                    }
                }
                G[s * (TT + 1) + t] = g;
            }
        }
        __syncthreads();
    }
}

__global__ void k_grad_px(const float* __restrict__ px, const float* __restrict__ al,
                          const float* __restrict__ occ, float* __restrict__ pxg) {
    size_t i = (size_t)blockIdx.x * blockDim.x + threadIdx.x;
    if (i >= (size_t)BB * SS * (TT + 1)) return;
    int t = (int)(i % (TT + 1));
    size_t r = i / (TT + 1);
    int s = (int)(r % SS), b = (int)(r / SS);
    const float* A = al + (size_t)b * SP1 * (TT + 1);
    float arg = A[s * (TT + 1) + t] + px[i] - A[(s + 1) * (TT + 1) + t];
    pxg[i] = occ[(size_t)b * SP1 * (TT + 1) + (s + 1) * (TT + 1) + t] * expf(fminf(arg, 30.0f));
}

__global__ void k_grad_py(const float* __restrict__ py, const float* __restrict__ al,
                          const float* __restrict__ occ, float* __restrict__ pyg) {
    size_t i = (size_t)blockIdx.x * blockDim.x + threadIdx.x;
    if (i >= (size_t)BB * SP1 * TT) return;
    int t = (int)(i % TT);
    size_t r = i / TT;
    int s = (int)(r % SP1), b = (int)(r / SP1);
    const float* A = al + (size_t)b * SP1 * (TT + 1);
    float arg = A[s * (TT + 1) + t] + py[i] - A[s * (TT + 1) + t + 1];
    pyg[i] = occ[(size_t)b * SP1 * (TT + 1) + s * (TT + 1) + t + 1] * expf(fminf(arg, 30.0f));
}

// ---------------- pruning ranges ----------------
__global__ void k_ranges_a(const float* __restrict__ pxg, const float* __restrict__ pyg,
                           int* __restrict__ sb) {
    int i = blockIdx.x * blockDim.x + threadIdx.x;
    if (i >= BB * TT) return;
    int b = i / TT, t = i % TT;
    float ring[RR + 1];
    float cs = 0.0f; ring[0] = 0.0f;
    float best = -3.0e38f; int bi = 0;
    for (int s = 0; s <= SS; ++s) {
        float tot = ((s < SS) ? pxg[((size_t)(b * SS + s)) * (TT + 1) + t] : 0.0f) +
                    pyg[((size_t)(b * SP1 + s)) * TT + t];
        cs += tot;
        int j = s + 1;
        float old = ring[(j >= RR) ? ((j - RR) % (RR + 1)) : 0];
        ring[j % (RR + 1)] = cs;
        if (j >= RR) {
            float diff = cs - old;
            if (diff > best) { best = diff; bi = j - RR; }
        }
    }
    sb[i] = bi;
}

__global__ void k_ranges_b(int* __restrict__ sb, const int* __restrict__ y,
                           int* __restrict__ ranges, int* __restrict__ psym) {
    int b = blockIdx.x;
    if (threadIdx.x != 0) return;
    int m = 0x7FFFFFFF;
    for (int t = 0; t < TT; ++t) {                 // cummin(s - (R-1)*t) + (R-1)*t
        int v = sb[b * TT + t] - (RR - 1) * t;
        if (v < m) m = v;
        sb[b * TT + t] = m + (RR - 1) * t;
    }
    int m2 = 0x7FFFFFFF;
    for (int t = TT - 1; t >= 0; --t) {            // reverse cummin, clip
        int v = sb[b * TT + t];
        if (v < m2) m2 = v;
        int s = m2;
        if (s < 0) s = 0;
        if (s > SP1 - RR) s = SP1 - RR;
        for (int r = 0; r < RR; ++r) {
            int row = s + r;
            ranges[(b * TT + t) * RR + r] = row;
            psym[(b * TT + t) * RR + r] = (row < SS) ? y[b * SS + row] : 0;
        }
    }
}

// ---------------- fused joiner GEMM + log-softmax gather (dominant stage) ----------------
__global__ void __launch_bounds__(32)
k_fused_joiner(const float* __restrict__ enc_p, const float* __restrict__ prd_p,
               const bf16_t* __restrict__ outw, const float* __restrict__ outb,
               const int* __restrict__ ranges, const int* __restrict__ psym,
               float* __restrict__ px2, float* __restrict__ py2) {
    __shared__ __align__(16) bf16_t tileB[16 * JJ];   // out_w tile (TDM target)
    __shared__ __align__(16) bf16_t tileA[16 * JJ];   // tanh(enc_p + prd_p) rows
    __shared__ float cl[16][16];
    __shared__ float btile[16];
    __shared__ int rb[16], rt[16], rs_[16], rp[16];

    int lane = threadIdx.x;
    int blk = blockIdx.x;

    if (lane < 16) {
        int g = blk * 16 + lane;                 // global row in (b,t,r) space
        int b = g / (TT * RR);
        int rem = g % (TT * RR);
        int t = rem / RR, r = rem % RR;
        rb[lane] = b; rt[lane] = t;
        rs_[lane] = ranges[(b * TT + t) * RR + r];
        rp[lane]  = psym[(b * TT + t) * RR + r];
    }
    __syncthreads();

    // build A tile = tanh(enc_p[b,t,:] + prd_p[b,srow,:]) in bf16
    for (int idx = lane; idx < 16 * JJ; idx += 32) {
        int row = idx >> 9, col = idx & (JJ - 1);
        float v = tanhf(enc_p[((size_t)(rb[row] * TT + rt[row])) * JJ + col] +
                        prd_p[((size_t)(rb[row] * SP1 + rs_[row])) * JJ + col]);
        tileA[idx] = f2b(v);
    }
    __syncthreads();

    v16bf af[16];
#pragma unroll
    for (int kc = 0; kc < 16; ++kc) af[kc] = load_frag_a(tileA, JJ, lane, kc * 32);

    float rm = -3.0e38f, rsum = 0.0f, vsym = NEGV, vblank = NEGV;  // lanes < 16
#if USE_TDM
    unsigned ldsB = (unsigned)(size_t)(void*)&tileB[0];
#endif

    for (int nt = 0; nt < NTILES_V; ++nt) {
        int n0 = nt * 16;
#if USE_TDM
        {
            unsigned long long ga = (unsigned long long)(size_t)(outw + (size_t)n0 * JJ);
            v4u g0;
            g0[0] = 1u;                                   // count=1, user descriptor
            g0[1] = ldsB;                                 // lds_addr
            g0[2] = (unsigned)(ga & 0xFFFFFFFFull);       // global_addr[31:0]
            g0[3] = (unsigned)((ga >> 32) & 0x01FFFFFFull) | (2u << 30);  // addr[56:32] | type=2
            v8i g1;
            g1[0] = (int)(1u << 16);                      // data_size=1 (2 bytes), wg_mask=0
            g1[1] = (int)(((unsigned)JJ) << 16);          // tensor_dim0 low16 @ bits63:48
            g1[2] = (int)(((unsigned)VP) << 16);          // td0 hi16=0, tensor_dim1 low16 @ 95:80
            g1[3] = (int)(((unsigned)JJ) << 16);          // td1 hi16=0, tile_dim0 @ 127:112
            g1[4] = 16;                                   // tile_dim1=16, tile_dim2=0
            g1[5] = (int)JJ;                              // tensor_dim0_stride low32
            g1[6] = 0;
            g1[7] = 0;
            asm volatile("tensor_load_to_lds %0, %1" :: "s"(g0), "s"(g1) : "memory");
            __builtin_amdgcn_s_wait_tensorcnt(0);
        }
        __syncthreads();
#else
        for (int idx = lane; idx < 16 * (JJ / 8); idx += 32) {
            int row = idx / (JJ / 8), c8 = idx % (JJ / 8);
            *(v8bf*)&tileB[row * JJ + c8 * 8] = *(const v8bf*)&outw[((size_t)(n0 + row)) * JJ + c8 * 8];
        }
        __syncthreads();
#endif
        if (lane < 16) btile[lane] = outb[n0 + lane];

        v8f c = {};
#pragma unroll
        for (int kc = 0; kc < 16; ++kc) {
            v16bf bfr = load_frag_b(tileB, JJ, lane, kc * 32);
            c = __builtin_amdgcn_wmma_f32_16x16x32_bf16(false, af[kc], false, bfr, (short)0, c, false, false);
        }
#pragma unroll
        for (int j = 0; j < 8; ++j) cl[j + ((lane >> 4) << 3)][lane & 15] = c[j];
        __syncthreads();

        if (lane < 16) {                           // lane owns row `lane`: online logsumexp
            int row = lane;
            float tm = -3.0e38f;
            float vals[16];
#pragma unroll
            for (int i = 0; i < 16; ++i) {
                int col = n0 + i;
                float v = (col < VV) ? (cl[row][i] + btile[i]) : NEGV;
                vals[i] = v;
                tm = fmaxf(tm, v);
                if (col == rp[row]) vsym = v;
                if (col == 0)       vblank = v;
            }
            float ts = 0.0f;
#pragma unroll
            for (int i = 0; i < 16; ++i) ts += expf(vals[i] - tm);
            float nm = fmaxf(rm, tm);
            rsum = rsum * expf(rm - nm) + ts * expf(tm - nm);
            rm = nm;
        }
        __syncthreads();
    }

    if (lane < 16) {
        float lse = rm + logf(rsum);
        int b = rb[lane], t = rt[lane], s = rs_[lane];
        if (s < SS) px2[((size_t)(b * SS + s)) * (TT + 1) + t] = vsym - lse;
        py2[((size_t)(b * SP1 + s)) * TT + t] = vblank - lse;
    }
}

__global__ void k_finalize(const float* __restrict__ loss, float* __restrict__ out) {
    if (blockIdx.x == 0 && threadIdx.x == 0) {
        float s = 0.0f;
        for (int b = 0; b < BB; ++b) s += loss[b];
        out[0] = -s / (float)BB;
    }
}

// ---------------- host side ----------------
extern "C" void kernel_launch(void* const* d_in, const int* in_sizes, int n_in,
                              void* d_out, int out_size, void* d_ws, size_t ws_size,
                              hipStream_t stream) {
    const float* encoder_out = (const float*)d_in[0];
    const int*   y        = (const int*)d_in[2];
    const float* pred_emb = (const float*)d_in[4];
    const float* am_w  = (const float*)d_in[5];
    const float* am_b  = (const float*)d_in[6];
    const float* am_ws = (const float*)d_in[7];
    const float* am_bs = (const float*)d_in[8];
    const float* lm_w  = (const float*)d_in[9];
    const float* lm_b  = (const float*)d_in[10];
    const float* lm_ws = (const float*)d_in[11];
    const float* lm_bs = (const float*)d_in[12];
    const float* enc_w = (const float*)d_in[13];
    const float* enc_b = (const float*)d_in[14];
    const float* prd_w = (const float*)d_in[15];
    const float* prd_b = (const float*)d_in[16];
    const float* out_w = (const float*)d_in[17];
    const float* out_b = (const float*)d_in[18];
    float* out = (float*)d_out;

    char* ws = (char*)d_ws;
    size_t off = 0;
    auto alloc = [&](size_t bytes) -> void* {
        void* p = ws + off;
        off = (off + bytes + 255) & ~(size_t)255;
        return p;
    };

    bf16_t* enc_bf  = (bf16_t*)alloc((size_t)MT * DD * 2);
    float*  pred_f  = (float*) alloc((size_t)MPREDP * DD * 4);
    bf16_t* pred_bf = (bf16_t*)alloc((size_t)MPREDP * DD * 2);
    bf16_t* amw_bf  = (bf16_t*)alloc((size_t)VP * DD * 2);
    bf16_t* lmw_bf  = (bf16_t*)alloc((size_t)VP * DD * 2);
    bf16_t* encw_bf = (bf16_t*)alloc((size_t)JJ * DD * 2);
    bf16_t* prdw_bf = (bf16_t*)alloc((size_t)JJ * DD * 2);
    bf16_t* outw_bf = (bf16_t*)alloc((size_t)VP * JJ * 2);
    float*  amb  = (float*)alloc((size_t)VP * 4);
    float*  lmb  = (float*)alloc((size_t)VP * 4);
    float*  encb = (float*)alloc((size_t)JJ * 4);
    float*  prdb = (float*)alloc((size_t)JJ * 4);
    float*  outbb= (float*)alloc((size_t)VP * 4);
    float*  am   = (float*)alloc((size_t)MT * VP * 4);
    float*  lm   = (float*)alloc((size_t)MPREDP * VP * 4);
    float*  amax = (float*)alloc((size_t)MT * 4);
    float*  lmax = (float*)alloc((size_t)MPRED * 4);
    bf16_t* e_am = (bf16_t*)alloc((size_t)MT * VP * 2);
    bf16_t* e_lm = (bf16_t*)alloc((size_t)BB * 80 * VP * 2);
    float*  nrm  = (float*)alloc((size_t)BB * 80 * TT * 4);
    float*  px   = (float*)alloc((size_t)BB * SS * (TT + 1) * 4);
    float*  py   = (float*)alloc((size_t)BB * SP1 * TT * 4);
    float*  alpha= (float*)alloc((size_t)BB * SP1 * (TT + 1) * 4);
    float*  occ  = (float*)alloc((size_t)BB * SP1 * (TT + 1) * 4);
    float*  pxg  = (float*)alloc((size_t)BB * SS * (TT + 1) * 4);
    float*  pyg  = (float*)alloc((size_t)BB * SP1 * TT * 4);
    int*    sb   = (int*)  alloc((size_t)BB * TT * 4);
    int*    rngs = (int*)  alloc((size_t)BB * TT * RR * 4);
    int*    psym = (int*)  alloc((size_t)BB * TT * RR * 4);
    float*  enc_p= (float*)alloc((size_t)MT * JJ * 4);
    float*  prd_p= (float*)alloc((size_t)MPREDP * JJ * 4);
    float*  px2  = (float*)alloc((size_t)BB * SS * (TT + 1) * 4);
    float*  py2  = (float*)alloc((size_t)BB * SP1 * TT * 4);
    float*  alpha2=(float*)alloc((size_t)BB * SP1 * (TT + 1) * 4);
    float*  loss1= (float*)alloc((size_t)BB * 4);
    float*  loss2= (float*)alloc((size_t)BB * 4);
    (void)ws_size; (void)in_sizes; (void)n_in; (void)out_size;

    auto blks = [](size_t n, int bs) { return (unsigned)((n + bs - 1) / bs); };

    // --- weight / input conversions (bf16, scaled, zero-padded) ---
    k_scale_w_bf<<<blks((size_t)VP * DD, 256), 256, 0, stream>>>(am_w, amw_bf, VV, VP, DD, am_ws);
    k_scale_w_bf<<<blks((size_t)VP * DD, 256), 256, 0, stream>>>(lm_w, lmw_bf, VV, VP, DD, lm_ws);
    k_scale_w_bf<<<blks((size_t)JJ * DD, 256), 256, 0, stream>>>(enc_w, encw_bf, JJ, JJ, DD, nullptr);
    k_scale_w_bf<<<blks((size_t)JJ * DD, 256), 256, 0, stream>>>(prd_w, prdw_bf, JJ, JJ, DD, nullptr);
    k_scale_w_bf<<<blks((size_t)VP * JJ, 256), 256, 0, stream>>>(out_w, outw_bf, VV, VP, JJ, nullptr);
    k_scale_bias<<<blks(VP, 256), 256, 0, stream>>>(am_b, amb, VV, VP, am_bs);
    k_scale_bias<<<blks(VP, 256), 256, 0, stream>>>(lm_b, lmb, VV, VP, lm_bs);
    k_scale_bias<<<blks(JJ, 256), 256, 0, stream>>>(enc_b, encb, JJ, JJ, nullptr);
    k_scale_bias<<<blks(JJ, 256), 256, 0, stream>>>(prd_b, prdb, JJ, JJ, nullptr);
    k_scale_bias<<<blks(VP, 256), 256, 0, stream>>>(out_b, outbb, VV, VP, nullptr);
    k_cvt_enc<<<blks((size_t)MT * DD, 256), 256, 0, stream>>>(encoder_out, enc_bf);
    k_gather_pred<<<blks((size_t)MPREDP * DD, 256), 256, 0, stream>>>(y, pred_emb, pred_f, pred_bf);

    // --- projections (WMMA GEMMs) ---
    k_gemm_bf16<<<dim3(MT / 16, VP / 16), 32, 0, stream>>>(enc_bf, amw_bf, amb, am, MT, VV, DD, DD, DD, VP);
    k_gemm_bf16<<<dim3(MPREDP / 16, VP / 16), 32, 0, stream>>>(pred_bf, lmw_bf, lmb, lm, MPRED, VV, DD, DD, DD, VP);
    k_gemm_bf16<<<dim3(MT / 16, JJ / 16), 32, 0, stream>>>(enc_bf, encw_bf, encb, enc_p, MT, JJ, DD, DD, DD, JJ);
    k_gemm_bf16<<<dim3(MPREDP / 16, JJ / 16), 32, 0, stream>>>(pred_bf, prdw_bf, prdb, prd_p, MPRED, JJ, DD, DD, DD, JJ);

    // --- simple px/py via WMMA einsum norm ---
    k_rowmax<<<MT, 32, 0, stream>>>(am, VP, VV, amax);
    k_rowmax<<<MPRED, 32, 0, stream>>>(lm, VP, VV, lmax);
    k_exp_am<<<blks((size_t)MT * VP, 256), 256, 0, stream>>>(am, amax, e_am);
    k_exp_lm<<<blks((size_t)BB * 80 * VP, 256), 256, 0, stream>>>(lm, lmax, e_lm);
    k_gemm_norm<<<dim3(80 / 16, TT / 16, BB), 32, 0, stream>>>(e_lm, e_am, lmax, amax, nrm);
    k_build_px<<<blks((size_t)BB * SS * (TT + 1), 256), 256, 0, stream>>>(am, lm, nrm, y, px);
    k_build_py<<<blks((size_t)BB * SP1 * TT, 256), 256, 0, stream>>>(am, lm, nrm, py);

    // --- lattice forward + gradient (occupancies) ---
    k_mi_forward<<<BB, 96, 0, stream>>>(px, py, alpha, loss1);
    k_mi_occ<<<BB, 96, 0, stream>>>(px, py, alpha, occ);
    k_grad_px<<<blks((size_t)BB * SS * (TT + 1), 256), 256, 0, stream>>>(px, alpha, occ, pxg);
    k_grad_py<<<blks((size_t)BB * SP1 * TT, 256), 256, 0, stream>>>(py, alpha, occ, pyg);

    // --- pruning ranges ---
    k_ranges_a<<<blks(BB * TT, 256), 256, 0, stream>>>(pxg, pyg, sb);
    k_ranges_b<<<BB, 32, 0, stream>>>(sb, y, rngs, psym);

    // --- fused joiner GEMM + log-softmax gather ---
    k_fill<<<blks((size_t)BB * SS * (TT + 1), 256), 256, 0, stream>>>(px2, (size_t)BB * SS * (TT + 1), NEGV);
    k_fill<<<blks((size_t)BB * SP1 * TT, 256), 256, 0, stream>>>(py2, (size_t)BB * SP1 * TT, NEGV);
    k_fused_joiner<<<(BB * TT * RR) / 16, 32, 0, stream>>>(enc_p, prd_p, outw_bf, outbb, rngs, psym, px2, py2);

    // --- final pruned lattice + loss ---
    k_mi_forward<<<BB, 96, 0, stream>>>(px2, py2, alpha2, loss2);
    k_finalize<<<1, 32, 0, stream>>>(loss2, out);
}
